// SAGE_87196426043909
// MI455X (gfx1250) — matrix operand order repaired and checked by reading, not compiled, not run on hardware
//
#include <hip/hip_runtime.h>

// ---------------------------------------------------------------------------
// GraphSAGE forward on MI455X (gfx1250): bf16 WMMA GEMMs, f32 accumulate,
// LDS-staged double-buffered B tiles shared by all 8 waves of a block.
// ---------------------------------------------------------------------------

constexpr int NN  = 20000;   // nodes
constexpr int NE  = 32000;   // edges
constexpr int F0  = 5120;    // input feat
constexpr int F1  = 1680;    // sage1 out
constexpr int F1P = 1696;    // F1 padded to multiple of 32 (53*32)
constexpr int F2  = 640;     // sage2 out
constexpr int F3  = 320;     // lin1 out
constexpr int F4  = 160;     // lin2 out
constexpr int WAVES_PER_BLOCK = 8;   // 256 threads, wave32

typedef __attribute__((ext_vector_type(16))) __bf16    v16bf;
typedef __attribute__((ext_vector_type(8)))  float     v8f;
typedef __attribute__((ext_vector_type(4)))  unsigned  v4u;

union FragU { v16bf v; v4u q[2]; };

__device__ __forceinline__ unsigned short f2bf(float f) {
  unsigned u = __float_as_uint(f);
  u += 0x7FFFu + ((u >> 16) & 1u);          // round to nearest even
  return (unsigned short)(u >> 16);
}
__device__ __forceinline__ float bf2f(unsigned short h) {
  return __uint_as_float(((unsigned)h) << 16);
}

// ---- elementwise converts / packing ---------------------------------------

__global__ void f32_to_bf16_k(const float* __restrict__ in,
                              unsigned short* __restrict__ out, size_t n) {
  size_t i = (size_t)blockIdx.x * blockDim.x + threadIdx.x;
  if (i < n) out[i] = f2bf(in[i]);
}

// W is KxN row-major f32; out is NxKP row-major bf16 (zero-padded K tail)
__global__ void pack_weight_k(const float* __restrict__ W,
                              unsigned short* __restrict__ out,
                              int K, int N, int KP) {
  size_t i = (size_t)blockIdx.x * blockDim.x + threadIdx.x;
  size_t total = (size_t)N * KP;
  if (i >= total) return;
  int n = (int)(i / KP);
  int k = (int)(i % KP);
  out[i] = (k < K) ? f2bf(W[(size_t)k * N + n]) : (unsigned short)0;
}

// ---- graph aggregation -----------------------------------------------------

__global__ void edge_count_k(const long long* __restrict__ ei,
                             float* __restrict__ cnt, int nE) {
  int e = blockIdx.x * blockDim.x + threadIdx.x;
  if (e >= nE) return;
  atomicAdd(&cnt[(int)ei[nE + e]], 1.0f);
}

__global__ void edge_agg_f32_k(const float* __restrict__ X,
                               const long long* __restrict__ ei,
                               float* __restrict__ acc, int C, int nE) {
  int e = blockIdx.x;
  int f = blockIdx.y * blockDim.x + threadIdx.x;
  if (f >= C) return;
  long long s = ei[e], d = ei[nE + e];
  atomicAdd(&acc[(size_t)d * C + f], X[(size_t)s * C + f]);
}

__global__ void edge_agg_bf16_k(const unsigned short* __restrict__ X,
                                const long long* __restrict__ ei,
                                float* __restrict__ acc, int C, int nE) {
  int e = blockIdx.x;
  int f = blockIdx.y * blockDim.x + threadIdx.x;
  if (f >= C) return;
  long long s = ei[e], d = ei[nE + e];
  atomicAdd(&acc[(size_t)d * C + f], bf2f(X[(size_t)s * C + f]));
}

// acc is NN x C f32 sums; divide by per-row count, emit bf16
__global__ void norm_bf16_k(const float* __restrict__ acc,
                            const float* __restrict__ cnt,
                            unsigned short* __restrict__ out,
                            size_t total, int C) {
  size_t i = (size_t)blockIdx.x * blockDim.x + threadIdx.x;
  if (i >= total) return;
  int row = (int)(i / C);
  out[i] = f2bf(acc[i] / fmaxf(cnt[row], 1.0f));
}

// ---- WMMA fragment loads ---------------------------------------------------
// A (MxK row-major, 16x32 bf16 tile): lane<16 holds K[0..7]+[16..23] of row
// lane&15; lane>=16 holds K[8..15]+[24..31].  (ISA 7.12.2 16-bit A layout)
__device__ __forceinline__ v16bf load_a_frag(const unsigned short* __restrict__ A,
                                             size_t rowOff, int k0, int lane) {
  int kb = (lane < 16) ? 0 : 8;
  FragU f;
  f.q[0] = *(const v4u*)(A + rowOff + k0 + kb);
  f.q[1] = *(const v4u*)(A + rowOff + k0 + 16 + kb);
  return f.v;
}

// B-fragment layout in LDS: [tile i][lane][16 bf16]  (32B per lane per tile)
__device__ __forceinline__ v16bf load_b_lds(const unsigned short* sm, int i, int lane) {
  FragU f;
  const v4u* p = (const v4u*)(sm + ((i * 32 + lane) * 16));
  f.q[0] = p[0];
  f.q[1] = p[1];
  return f.v;
}

// Cooperative stage of NT B tiles (32x16 bf16 each) for k-step k0 into LDS.
// Bt is NxK row-major; lane fragment b[8h+j] = Bt[col][k0+kb+16h+j], kb = lane<16?0:8.
template <int NT>
__device__ __forceinline__ void stage_b(const unsigned short* __restrict__ Bt,
                                        unsigned short* sm,
                                        int nt0, int ntiles, int k0, int K) {
  for (int t = threadIdx.x; t < NT * 64; t += 256) {
    int i = t >> 6;          // tile index 0..NT-1
    int l = (t >> 1) & 31;   // lane
    int h = t & 1;           // 16-byte half
    int nt = nt0 + i; if (nt >= ntiles) nt = ntiles - 1;   // clamp (store guarded later)
    int col = nt * 16 + (l & 15);
    int kb  = (l < 16) ? 0 : 8;
    v4u d = *(const v4u*)(Bt + (size_t)col * K + k0 + kb + 16 * h);
    *(v4u*)(sm + ((i * 32 + l) * 16 + 8 * h)) = d;
  }
}

// ---- fused GEMM: out = act(A1@B1t' + A2@B2t' + bias) ----------------------
// MODE 1: bias + ReLU.   MODE 2: bias + ReLU + BatchNorm(eval).
template <int NT, int MODE>
__global__ __launch_bounds__(256) void gemm_bf16_wmma(
    const unsigned short* __restrict__ A1, const unsigned short* __restrict__ B1t,
    const unsigned short* __restrict__ A2, const unsigned short* __restrict__ B2t,
    const float* __restrict__ bias,
    const float* __restrict__ bn_g, const float* __restrict__ bn_b,
    const float* __restrict__ bn_m, const float* __restrict__ bn_v,
    unsigned short* __restrict__ outb,
    int M, int N, int K, int ldOut) {
  __shared__ unsigned short smB[2][2][NT * 32 * 16];   // [buf][stream][frag data]

  const int lane = threadIdx.x & 31;
  const int wave = threadIdx.x >> 5;
  const int mt = blockIdx.y * WAVES_PER_BLOCK + wave;
  const int ntiles = N >> 4;
  const int nt0 = blockIdx.x * NT;
  const int col16 = lane & 15;
  const bool mvalid = (mt * 16 < M);
  const int mtc = mvalid ? mt : (M / 16 - 1);            // clamp; stores guarded
  const size_t rowOff = (size_t)(mtc * 16 + col16) * K;  // A row for this lane

  v8f acc[NT] = {};

  // prologue: stage k0 = 0
  stage_b<NT>(B1t, smB[0][0], nt0, ntiles, 0, K);
  if (A2 != nullptr) stage_b<NT>(B2t, smB[0][1], nt0, ntiles, 0, K);
  __syncthreads();

  for (int k0 = 0; k0 < K; k0 += 32) {
    const int cur = (k0 >> 5) & 1;
    const int nxt = cur ^ 1;

    // stage next k-step's B tiles into the other buffer (loads overlap WMMA)
    if (k0 + 32 < K) {
      stage_b<NT>(B1t, smB[nxt][0], nt0, ntiles, k0 + 32, K);
      if (A2 != nullptr) stage_b<NT>(B2t, smB[nxt][1], nt0, ntiles, k0 + 32, K);
      __builtin_prefetch(A1 + rowOff + k0 + 32, 0, 0);   // global_prefetch_b8
    }

    // compute on current buffer
    v16bf a = load_a_frag(A1, rowOff, k0, lane);
#pragma unroll
    for (int i = 0; i < NT; ++i) {
      v16bf b = load_b_lds(smB[cur][0], i, lane);
      acc[i] = __builtin_amdgcn_wmma_f32_16x16x32_bf16(
          false, a, false, b, (short)0, acc[i], false, false);
    }
    if (A2 != nullptr) {
      v16bf a2 = load_a_frag(A2, rowOff, k0, lane);
#pragma unroll
      for (int i = 0; i < NT; ++i) {
        v16bf b = load_b_lds(smB[cur][1], i, lane);
        acc[i] = __builtin_amdgcn_wmma_f32_16x16x32_bf16(
            false, a2, false, b, (short)0, acc[i], false, false);
      }
    }
    __syncthreads();   // cur fully consumed; nxt fully written & visible
  }

  if (!mvalid) return;

  // C/D layout: c[v] = C[mt*16 + (lane<16 ? v : 8+v)][nt*16 + (lane&15)]
  const int rbase = mt * 16 + ((lane < 16) ? 0 : 8);
#pragma unroll
  for (int i = 0; i < NT; ++i) {
    int nt = nt0 + i;
    if (nt >= ntiles) break;
    int col = nt * 16 + col16;
    float bv = bias[col];
    float g = 0.f, be = 0.f, mu = 0.f, iv = 0.f;
    if (MODE == 2) {
      g = bn_g[col]; be = bn_b[col]; mu = bn_m[col];
      iv = rsqrtf(bn_v[col] + 1e-5f);
    }
#pragma unroll
    for (int v = 0; v < 8; ++v) {
      float x = acc[i][v] + bv;
      x = fmaxf(x, 0.0f);                         // ReLU
      if (MODE == 2) x = (x - mu) * iv * g + be;  // BatchNorm (eval)
      outb[(size_t)(rbase + v) * ldOut + col] = f2bf(x);
    }
  }
}

// ---- final tiny layer: (20000x160) @ (160x2) + b, f32 out ------------------
__global__ void lin3_k(const unsigned short* __restrict__ o2,
                       const float* __restrict__ W, const float* __restrict__ b,
                       float* __restrict__ out, int M) {
  int row = blockIdx.x * blockDim.x + threadIdx.x;
  if (row >= M) return;
  float a0 = b[0], a1 = b[1];
  const unsigned short* p = o2 + (size_t)row * F4;
  for (int k = 0; k < F4; ++k) {
    float v = bf2f(p[k]);
    a0 += v * W[k * 2 + 0];
    a1 += v * W[k * 2 + 1];
  }
  out[row * 2 + 0] = a0;
  out[row * 2 + 1] = a1;
}

// ---------------------------------------------------------------------------

extern "C" void kernel_launch(void* const* d_in, const int* in_sizes, int n_in,
                              void* d_out, int out_size, void* d_ws, size_t ws_size,
                              hipStream_t stream) {
  const float* x    = (const float*)d_in[0];
  const float* W1l  = (const float*)d_in[1];
  const float* b1   = (const float*)d_in[2];
  const float* W1r  = (const float*)d_in[3];
  const float* W2l  = (const float*)d_in[4];
  const float* b2   = (const float*)d_in[5];
  const float* W2r  = (const float*)d_in[6];
  const float* bn_g = (const float*)d_in[7];
  const float* bn_b = (const float*)d_in[8];
  const float* bn_m = (const float*)d_in[9];
  const float* bn_v = (const float*)d_in[10];
  const float* l1W  = (const float*)d_in[11];
  const float* l1b  = (const float*)d_in[12];
  const float* l2W  = (const float*)d_in[13];
  const float* l2b  = (const float*)d_in[14];
  const float* l3W  = (const float*)d_in[15];
  const float* l3b  = (const float*)d_in[16];
  const long long* ei = (const long long*)d_in[17];
  float* out = (float*)d_out;
  (void)in_sizes; (void)n_in; (void)out_size; (void)ws_size;

  char* ws = (char*)d_ws;
  size_t off = 0;
  auto alloc = [&](size_t bytes) -> char* {
    char* p = ws + off;
    off = (off + bytes + 255) & ~(size_t)255;
    return p;
  };
  unsigned short* xb    = (unsigned short*)alloc((size_t)NN * F0 * 2);
  float*          aggx  = (float*)         alloc((size_t)NN * F0 * 4);
  unsigned short* aggxb = (unsigned short*)alloc((size_t)NN * F0 * 2);
  float*          cnt   = (float*)         alloc((size_t)NN * 4);
  unsigned short* W1lt  = (unsigned short*)alloc((size_t)F1 * F0 * 2);
  unsigned short* W1rt  = (unsigned short*)alloc((size_t)F1 * F0 * 2);
  unsigned short* W2lt  = (unsigned short*)alloc((size_t)F2 * F1P * 2);
  unsigned short* W2rt  = (unsigned short*)alloc((size_t)F2 * F1P * 2);
  unsigned short* l1Wt  = (unsigned short*)alloc((size_t)F3 * F2 * 2);
  unsigned short* l2Wt  = (unsigned short*)alloc((size_t)F4 * F3 * 2);
  unsigned short* hb    = (unsigned short*)alloc((size_t)NN * F1P * 2);
  float*          aggh  = (float*)         alloc((size_t)NN * F1P * 4);
  unsigned short* agghb = (unsigned short*)alloc((size_t)NN * F1P * 2);
  unsigned short* ob    = (unsigned short*)alloc((size_t)NN * F2 * 2);
  unsigned short* o1b   = (unsigned short*)alloc((size_t)NN * F3 * 2);
  unsigned short* o2b   = (unsigned short*)alloc((size_t)NN * F4 * 2);

  // zero accumulators / padded buffers (every call; graph-capture safe)
  hipMemsetAsync(aggx, 0, (size_t)NN * F0 * 4, stream);
  hipMemsetAsync(cnt,  0, (size_t)NN * 4, stream);
  hipMemsetAsync(hb,   0, (size_t)NN * F1P * 2, stream);
  hipMemsetAsync(aggh, 0, (size_t)NN * F1P * 4, stream);

  // converts + weight packing
  { size_t n = (size_t)NN * F0;
    f32_to_bf16_k<<<dim3((unsigned)((n + 255) / 256)), 256, 0, stream>>>(x, xb, n); }
  { size_t n = (size_t)F1 * F0;
    pack_weight_k<<<dim3((unsigned)((n + 255) / 256)), 256, 0, stream>>>(W1l, W1lt, F0, F1, F0);
    pack_weight_k<<<dim3((unsigned)((n + 255) / 256)), 256, 0, stream>>>(W1r, W1rt, F0, F1, F0); }
  { size_t n = (size_t)F2 * F1P;
    pack_weight_k<<<dim3((unsigned)((n + 255) / 256)), 256, 0, stream>>>(W2l, W2lt, F1, F2, F1P);
    pack_weight_k<<<dim3((unsigned)((n + 255) / 256)), 256, 0, stream>>>(W2r, W2rt, F1, F2, F1P); }
  { size_t n = (size_t)F3 * F2;
    pack_weight_k<<<dim3((unsigned)((n + 255) / 256)), 256, 0, stream>>>(l1W, l1Wt, F2, F3, F2); }
  { size_t n = (size_t)F4 * F3;
    pack_weight_k<<<dim3((unsigned)((n + 255) / 256)), 256, 0, stream>>>(l2W, l2Wt, F3, F4, F3); }

  // graph mean-aggregation of x
  edge_count_k<<<dim3((NE + 255) / 256), 256, 0, stream>>>(ei, cnt, NE);
  edge_agg_f32_k<<<dim3(NE, (F0 + 255) / 256), 256, 0, stream>>>(x, ei, aggx, F0, NE);
  { size_t n = (size_t)NN * F0;
    norm_bf16_k<<<dim3((unsigned)((n + 255) / 256)), 256, 0, stream>>>(aggx, cnt, aggxb, n, F0); }

  const int MT = (NN / 16 + WAVES_PER_BLOCK - 1) / WAVES_PER_BLOCK;  // 157

  // SAGE layer 1: h = BN(ReLU(agg@W1l + b1 + x@W1r))  -> hb (stride F1P)
  gemm_bf16_wmma<4, 2><<<dim3((F1 / 16 + 3) / 4, MT), 256, 0, stream>>>(
      aggxb, W1lt, xb, W1rt, b1, bn_g, bn_b, bn_m, bn_v, hb, NN, F1, F0, F1P);

  // graph mean-aggregation of h (padded cols are zeros -> harmless)
  edge_agg_bf16_k<<<dim3(NE, (F1P + 255) / 256), 256, 0, stream>>>(hb, ei, aggh, F1P, NE);
  { size_t n = (size_t)NN * F1P;
    norm_bf16_k<<<dim3((unsigned)((n + 255) / 256)), 256, 0, stream>>>(aggh, cnt, agghb, n, F1P); }

  // SAGE layer 2: o = ReLU(agg@W2l + b2 + h@W2r)
  gemm_bf16_wmma<4, 1><<<dim3((F2 / 16 + 3) / 4, MT), 256, 0, stream>>>(
      agghb, W2lt, hb, W2rt, b2, nullptr, nullptr, nullptr, nullptr, ob, NN, F2, F1P, F2);

  // lin1 + ReLU
  gemm_bf16_wmma<4, 1><<<dim3((F3 / 16 + 3) / 4, MT), 256, 0, stream>>>(
      ob, l1Wt, nullptr, nullptr, l1b, nullptr, nullptr, nullptr, nullptr, o1b, NN, F3, F2, F3);

  // lin2 + ReLU
  gemm_bf16_wmma<4, 1><<<dim3((F4 / 16 + 3) / 4, MT), 256, 0, stream>>>(
      o1b, l2Wt, nullptr, nullptr, l2b, nullptr, nullptr, nullptr, nullptr, o2b, NN, F4, F3, F4);

  // lin3 (160 -> 2), f32 output
  lin3_k<<<dim3((NN + 255) / 256), 256, 0, stream>>>(o2b, l3W, l3b, out, NN);
}
